// LASAGE_74998718923050
// MI455X (gfx1250) — matrix operand order repaired and checked by reading, not compiled, not run on hardware
//
#include <hip/hip_runtime.h>
#include <math.h>

// ---------------- problem constants (match reference) ----------------
#define NN      50000
#define FDIM    512
#define CONCAT  4
#define HID     8
#define NCLS    7
#define EDGES   1600000
#define ALPHA   0.2f
#define CH      (CONCAT * HID)   // 32 combined hidden width

typedef __attribute__((ext_vector_type(16))) _Float16 v16h;
typedef __attribute__((ext_vector_type(8)))  float    v8f;
typedef __attribute__((ext_vector_type(4)))  float    f32x4;

// ---------------------------------------------------------------------
// Phase A: fused projection  P = x @ [Wl1 ; Wr1]^T  per branch, via WMMA.
// One wave per (16-node tile, branch). A = x tile [16 x 512], B = weight
// tile [512 x 16] (cols 0..7 = lin_l rows, 8..15 = lin_r rows), f32->f16
// convert on the fly, f32 accumulate through v_wmma_f32_16x16x32_f16.
// x is streamed with non-temporal b128 loads (read-once, keep L2 for the
// edge-phase tables); weights go through the cached path.
// ---------------------------------------------------------------------
__global__ __launch_bounds__(256) void lasage_proj_wmma(
    const float* __restrict__ x,    // [C, N, F]
    const float* __restrict__ Wl1,  // [C, HID, F]
    const float* __restrict__ Wr1,  // [C, HID, F]
    float* __restrict__ YL,         // [N, CH]
    float* __restrict__ YR)         // [N, CH]
{
    const int ntiles = NN / 16;                       // 3125 (exact)
    const int gwave  = blockIdx.x * 8 + (threadIdx.x >> 5);
    if (gwave >= ntiles * CONCAT) return;             // uniform per wave

    const int c    = gwave / ntiles;
    const int tile = gwave % ntiles;
    const int base = tile * 16;

    const int lane = threadIdx.x & 31;
    const int h    = lane >> 4;                       // lane-half
    const int l    = lane & 15;

    // A-matrix row for this lane: node (base + l), branch c
    const float* __restrict__ xrow = x + ((size_t)c * NN + (base + l)) * FDIM;
    // B-matrix column j = l: rows of Wl1 (j<8) or Wr1 (j>=8)
    const float* __restrict__ wrow = (l < 8)
        ? (Wl1 + ((size_t)c * HID + l)       * FDIM)
        : (Wr1 + ((size_t)c * HID + (l - 8)) * FDIM);

    // Resolve the YL/YR split ONCE into a base pointer (uniform control flow
    // afterwards -> straight-line coalesced stores, no exec-mask juggling).
    float* __restrict__ dstbase = (l < 8)
        ? (YL + c * HID + l)
        : (YR + c * HID + (l - 8));

    v8f acc = {};
#pragma unroll 2
    for (int kstep = 0; kstep < FDIM / 32; ++kstep) {
        const int k0 = kstep * 32;

        // A layout (16-bit 16x32): elems 0..7 -> K=k0+h*8+i ; 8..15 -> K=k0+16+h*8+i
        const f32x4 xa0 = __builtin_nontemporal_load((const f32x4*)(xrow + k0 + h * 8));
        const f32x4 xa1 = __builtin_nontemporal_load((const f32x4*)(xrow + k0 + h * 8 + 4));
        const f32x4 xa2 = __builtin_nontemporal_load((const f32x4*)(xrow + k0 + 16 + h * 8));
        const f32x4 xa3 = __builtin_nontemporal_load((const f32x4*)(xrow + k0 + 16 + h * 8 + 4));
        // B layout (16-bit 32x16): elems 0..7 -> K=k0+h*8+i ; 8..15 -> K=k0+16+h*8+i, column = l
        const f32x4 wb0 = *(const f32x4*)(wrow + k0 + h * 8);
        const f32x4 wb1 = *(const f32x4*)(wrow + k0 + h * 8 + 4);
        const f32x4 wb2 = *(const f32x4*)(wrow + k0 + 16 + h * 8);
        const f32x4 wb3 = *(const f32x4*)(wrow + k0 + 16 + h * 8 + 4);

        v16h a, b;
#pragma unroll
        for (int i = 0; i < 4; ++i) {
            a[i]      = (_Float16)xa0[i];
            a[i + 4]  = (_Float16)xa1[i];
            a[i + 8]  = (_Float16)xa2[i];
            a[i + 12] = (_Float16)xa3[i];
            b[i]      = (_Float16)wb0[i];
            b[i + 4]  = (_Float16)wb1[i];
            b[i + 8]  = (_Float16)wb2[i];
            b[i + 12] = (_Float16)wb3[i];
        }
        acc = __builtin_amdgcn_wmma_f32_16x16x32_f16(
            /*neg_a=*/false, a, /*neg_b=*/false, b,
            /*c_mod=*/(short)0, acc, /*reuse_a=*/false, /*reuse_b=*/false);
    }

    // D layout: VGPR r -> M = r + 8*h (node row), N = l (output column)
#pragma unroll
    for (int r = 0; r < 8; ++r) {
        const int node = base + r + (h ? 8 : 0);
        dstbase[(size_t)node * CH] = acc[r];
    }
}

// ---------------------------------------------------------------------
// Zero helper (graph-capture safe; no hipMemset on stream needed)
// ---------------------------------------------------------------------
__global__ void lasage_zero(float* __restrict__ p, int n)
{
    int t = blockIdx.x * blockDim.x + threadIdx.x;
    if (t < n) p[t] = 0.0f;
}

// ---------------------------------------------------------------------
// Edge scatter: one wave per edge, lane = feature channel (32 = one 128B
// row). AGG[dst][j] += SRCTAB[src][j]; optionally count degree. Both the
// gather table and the atomic target are ~6.4 MB -> L2-resident.
// ---------------------------------------------------------------------
__global__ void lasage_edge_scatter(
    const int* __restrict__ esrc, const int* __restrict__ edst,
    const float* __restrict__ srctab, float* __restrict__ agg,
    float* __restrict__ deg, int count_deg)
{
    long long t = (long long)blockIdx.x * blockDim.x + threadIdx.x;
    long long e = t >> 5;
    int j = (int)(t & 31);
    if (e >= EDGES) return;
    int s = esrc[e];
    int d = edst[e];
    atomicAdd(&agg[(size_t)d * CH + j], srctab[(size_t)s * CH + j]);
    if (count_deg && j == 0) atomicAdd(&deg[d], 1.0f);
}

// ---------------------------------------------------------------------
// Phase D: x2 = elu( AGG/clip(deg,1) + bl1 + YR , alpha=0.2 )
// ---------------------------------------------------------------------
__global__ void lasage_hidden(
    const float* __restrict__ agg, const float* __restrict__ yr,
    const float* __restrict__ bl1, const float* __restrict__ deg,
    float* __restrict__ x2)
{
    int t = blockIdx.x * blockDim.x + threadIdx.x;   // over N*CH
    if (t >= NN * CH) return;
    int n = t >> 5;
    int j = t & 31;                                   // j == c*HID + h, matches bl1 flat
    float dg = deg[n];
    dg = dg < 1.0f ? 1.0f : dg;
    float v = agg[t] / dg + bl1[j] + yr[t];
    v = v > 0.0f ? v : ALPHA * (__expf(v) - 1.0f);
    x2[t] = v;
}

// ---------------------------------------------------------------------
// Phase F: out[n] = (AGG2[n]/clip(deg,1)) @ Wl2^T + bl2 + x2[n] @ Wr2^T
// ---------------------------------------------------------------------
__global__ void lasage_output(
    const float* __restrict__ agg2, const float* __restrict__ x2,
    const float* __restrict__ deg,  const float* __restrict__ Wl2,
    const float* __restrict__ bl2,  const float* __restrict__ Wr2,
    float* __restrict__ out)
{
    int n = blockIdx.x * blockDim.x + threadIdx.x;
    if (n >= NN) return;
    float dg = deg[n];
    dg = dg < 1.0f ? 1.0f : dg;
    const float inv = 1.0f / dg;
    const float* a  = agg2 + (size_t)n * CH;
    const float* xx = x2   + (size_t)n * CH;
#pragma unroll
    for (int k = 0; k < NCLS; ++k) {
        float s = bl2[k];
#pragma unroll
        for (int j = 0; j < CH; ++j)
            s += a[j] * inv * Wl2[k * CH + j] + xx[j] * Wr2[k * CH + j];
        out[(size_t)n * NCLS + k] = s;
    }
}

// ---------------------------------------------------------------------
extern "C" void kernel_launch(void* const* d_in, const int* in_sizes, int n_in,
                              void* d_out, int out_size, void* d_ws, size_t ws_size,
                              hipStream_t stream)
{
    (void)in_sizes; (void)n_in; (void)out_size; (void)ws_size;

    const float* x    = (const float*)d_in[0];   // [C, N, F]
    const int*   ei   = (const int*)  d_in[1];   // [2, E]
    const float* Wl1  = (const float*)d_in[2];   // [C, HID, F]
    const float* bl1  = (const float*)d_in[3];   // [C, HID]
    const float* Wr1  = (const float*)d_in[4];   // [C, HID, F]
    const float* Wl2  = (const float*)d_in[5];   // [NCLS, CH]
    const float* bl2  = (const float*)d_in[6];   // [NCLS]
    const float* Wr2  = (const float*)d_in[7];   // [NCLS, CH]
    float*       out  = (float*)d_out;           // [N, NCLS]

    const int* esrc = ei;
    const int* edst = ei + EDGES;

    // workspace layout (floats): YL | YR | AGG | X2 | AGG2 | deg  (~32.2 MB)
    float* ws   = (float*)d_ws;
    float* YL   = ws;
    float* YR   = YL   + (size_t)NN * CH;
    float* AGG  = YR   + (size_t)NN * CH;
    float* X2   = AGG  + (size_t)NN * CH;
    float* AGG2 = X2   + (size_t)NN * CH;
    float* DEG  = AGG2 + (size_t)NN * CH;

    const int nch = NN * CH;

    // zero accumulators + degrees
    lasage_zero<<<(nch + 255) / 256, 256, 0, stream>>>(AGG,  nch);
    lasage_zero<<<(nch + 255) / 256, 256, 0, stream>>>(AGG2, nch);
    lasage_zero<<<(NN  + 255) / 256, 256, 0, stream>>>(DEG,  NN);

    // Phase A: WMMA projection (3125 tiles x 4 branches = 12500 waves, 8/block)
    {
        const int waves  = (NN / 16) * CONCAT;
        const int blocks = (waves + 7) / 8;
        lasage_proj_wmma<<<blocks, 256, 0, stream>>>(x, Wl1, Wr1, YL, YR);
    }

    // Phase C: aggregate lin_l projections + degree count
    {
        const long long thr = (long long)EDGES * 32;
        const int blocks = (int)((thr + 255) / 256);
        lasage_edge_scatter<<<blocks, 256, 0, stream>>>(esrc, edst, YL, AGG, DEG, 1);
    }

    // Phase D: hidden = elu(agg/deg + bl1 + yr)
    lasage_hidden<<<(nch + 255) / 256, 256, 0, stream>>>(AGG, YR, bl1, DEG, X2);

    // Phase E: second aggregation of x2
    {
        const long long thr = (long long)EDGES * 32;
        const int blocks = (int)((thr + 255) / 256);
        lasage_edge_scatter<<<blocks, 256, 0, stream>>>(esrc, edst, X2, AGG2, DEG, 0);
    }

    // Phase F: output layer
    lasage_output<<<(NN + 255) / 256, 256, 0, stream>>>(AGG2, X2, DEG, Wl2, bl2, Wr2, out);
}
